// IDGatedLoRA_65412351918160
// MI455X (gfx1250) — matrix operand order repaired and checked by reading, not compiled, not run on hardware
//
#include <hip/hip_runtime.h>

// ---- problem constants (match reference) ----
constexpr int N_TASKS = 8;
constexpr int IN_DIM  = 4096;
constexpr int OUT_DIM = 4096;
constexpr int RANK    = 16;
constexpr int BATCH   = 16384;
constexpr int RT      = N_TASKS * RANK;            // 128 = folded task*rank axis

constexpr size_t AWS_ELEMS = (size_t)N_TASKS * RANK * IN_DIM;  // 524288
constexpr size_t BWS_ELEMS = (size_t)OUT_DIM * RT;             // 524288
constexpr size_t WS_NEEDED = (AWS_ELEMS + BWS_ELEMS) * 2;      // 2 MiB

typedef __attribute__((ext_vector_type(16))) __bf16 v16bf;
typedef __attribute__((ext_vector_type(8)))  __bf16 v8bf;
typedef __attribute__((ext_vector_type(2)))  __bf16 v2bf;
typedef __attribute__((ext_vector_type(8)))  float  v8f;
typedef __attribute__((ext_vector_type(4)))  float  f32x4;
typedef __attribute__((ext_vector_type(4)))  unsigned u32x4;

// round-to-nearest-even f32 -> bf16
__device__ __forceinline__ __bf16 f2bf(float f) {
    unsigned u = __builtin_bit_cast(unsigned, f);
    u += 0x7FFFu + ((u >> 16) & 1u);
    unsigned short h = (unsigned short)(u >> 16);
    return __builtin_bit_cast(__bf16, h);
}
__device__ __forceinline__ unsigned pack2bf(float lo, float hi) {
#if __has_builtin(__builtin_amdgcn_cvt_pk_bf16_f32)
    v2bf p = __builtin_amdgcn_cvt_pk_bf16_f32(lo, hi);
    return __builtin_bit_cast(unsigned, p);
#else
    unsigned a = (unsigned)__builtin_bit_cast(unsigned short, f2bf(lo));
    unsigned b = (unsigned)__builtin_bit_cast(unsigned short, f2bf(hi));
    return a | (b << 16);
#endif
}
__device__ __forceinline__ v16bf cat8(v8bf lo, v8bf hi) {
    return __builtin_shufflevector(lo, hi, 0,1,2,3,4,5,6,7,8,9,10,11,12,13,14,15);
}

// ---------------- preprocess: fp32 -> bf16, operand-major layouts ----------------
// A_ws[t][r][i] = A[t][i][r]   (stage-1 WMMA B-operand: two runs of 8 contiguous K along i)
__global__ __launch_bounds__(256)
void preconv_A(const float* __restrict__ A, __bf16* __restrict__ Aws) {
    int idx = blockIdx.x * 256 + threadIdx.x;        // over T*R*IN
    int i  = idx & (IN_DIM - 1);
    int tr = idx >> 12;                              // t*16 + r
    int r  = tr & (RANK - 1);
    int t  = tr >> 4;
    Aws[idx] = f2bf(A[((size_t)t * IN_DIM + i) * RANK + r]);
}
// B_ws[o][r'] = B_flat[r'][o]  (stage-2 WMMA B-operand: two runs of 8 contiguous K along r')
__global__ __launch_bounds__(256)
void preconv_B(const float* __restrict__ B, __bf16* __restrict__ Bws) {
    int idx = blockIdx.x * 256 + threadIdx.x;        // over RT*OUT, o fastest (coalesced read)
    int o  = idx & (OUT_DIM - 1);
    int rp = idx >> 12;
    Bws[(size_t)o * RT + rp] = f2bf(B[idx]);
}

// ---------------- fused routed-LoRA kernel ----------------
// One workgroup = 32 batch rows (two 16-row WMMA tiles); 8 waves. Every A/B
// operand load feeds TWO WMMAs (one per row tile) => half the L2 traffic.
template <bool PRE>
__global__ __launch_bounds__(256)
void id_gated_lora_kernel(const float* __restrict__ x,
                          const float* __restrict__ lora_A,   // [T, IN, R]
                          const float* __restrict__ lora_B,   // [T, R, OUT] == B_flat[128, OUT]
                          const int*   __restrict__ task_id,
                          const __bf16* __restrict__ Aws,     // [T*R][IN]  (PRE only)
                          const __bf16* __restrict__ Bws,     // [OUT][128] (PRE only)
                          float*       __restrict__ out) {
    constexpr int XSTR = 72;    // bf16 stride: 16B-aligned rows, conflict-free b128
    constexpr int GSTR = 136;   // bf16 stride for gated xA

    __shared__ __bf16 Xs[32][XSTR];     // 32 x 64 bf16 chunk of x
    __shared__ __bf16 xAg[32][GSTR];    // gated xA, [32 rows][128]
    __shared__ int    tids[32];

    const int tid  = threadIdx.x;
    const int lane = tid & 31;
    const int wave = tid >> 5;
    const int row0 = blockIdx.x * 32;

    if (tid < 32) tids[tid] = task_id[row0 + tid];

    const int M    = lane & 15;             // A-operand row (within a 16-row tile)
    const int kb8  = (lane >> 4) << 3;      // A-operand K sub-offset (0 / 8)
    const int nn   = lane & 15;             // B-operand column
    const int ko16 = (lane >> 4) << 3;      // B-operand K sub-offset (0 / 8)
    const int Mb   = (lane >> 4) << 3;      // C/D layout row base

    const __bf16* arow = PRE ? (Aws + (size_t)(wave * RANK + nn) * IN_DIM) : nullptr;
    const float*  Araw = lora_A + (size_t)wave * IN_DIM * RANK;

    // ---------------- Stage 1: S_w = X_tile[32x4096] @ A[w][4096x16] ----------------
    v8f acc0 = {}, acc1 = {};
    for (int kk = 0; kk < IN_DIM; kk += 64) {
        { // cooperative stage+convert: 32 rows x 64 cols; 8 f32 -> 8 bf16 per thread
            int lin = tid << 3;
            int r   = lin >> 6;
            int c   = lin & 63;
            const float* xp = x + (size_t)(row0 + r) * IN_DIM + kk + c;
            f32x4 v0 = *(const f32x4*)xp;
            f32x4 v1 = *(const f32x4*)(xp + 4);
            u32x4 p;
            p.x = pack2bf(v0.x, v0.y);
            p.y = pack2bf(v0.z, v0.w);
            p.z = pack2bf(v1.x, v1.y);
            p.w = pack2bf(v1.z, v1.w);
            *(u32x4*)&Xs[r][c] = p;
        }
        __syncthreads();

        #pragma unroll
        for (int k32 = 0; k32 < 64; k32 += 32) {
            v16bf b;
            const int Kst = kk + k32 + ko16;
            if constexpr (PRE) {
                v8bf blo = *(const v8bf*)(arow + Kst);
                v8bf bhi = *(const v8bf*)(arow + Kst + 16);
                b = cat8(blo, bhi);
            } else {
                #pragma unroll
                for (int e = 0; e < 16; ++e)
                    b[e] = f2bf(Araw[(size_t)(Kst + e + (e & 8)) * RANK + nn]);
            }
            // two row tiles share one B operand
            v16bf a0 = cat8(*(const v8bf*)&Xs[M][k32 + kb8],
                            *(const v8bf*)&Xs[M][k32 + 16 + kb8]);
            v16bf a1 = cat8(*(const v8bf*)&Xs[16 + M][k32 + kb8],
                            *(const v8bf*)&Xs[16 + M][k32 + 16 + kb8]);
            acc0 = __builtin_amdgcn_wmma_f32_16x16x32_bf16(
                false, a0, false, b, (short)0, acc0, false, false);
            acc1 = __builtin_amdgcn_wmma_f32_16x16x32_bf16(
                false, a1, false, b, (short)0, acc1, false, false);
        }
        __syncthreads();
    }

    // ---------------- gate by task id, deposit bf16 xA_g ----------------
    #pragma unroll
    for (int r = 0; r < 8; ++r) {
        int m0 = Mb + r, m1 = 16 + Mb + r;
        xAg[m0][wave * 16 + nn] = f2bf((tids[m0] == wave) ? acc0[r] : 0.0f);
        xAg[m1][wave * 16 + nn] = f2bf((tids[m1] == wave) ? acc1[r] : 0.0f);
    }
    __syncthreads();

    // ---------------- Stage 2: out_tile = xA_g[32x128] @ B_flat[128x4096] ----------------
    v16bf a2[2][4];
    #pragma unroll
    for (int c = 0; c < 4; ++c) {
        a2[0][c] = cat8(*(const v8bf*)&xAg[M][c * 32 + kb8],
                        *(const v8bf*)&xAg[M][c * 32 + 16 + kb8]);
        a2[1][c] = cat8(*(const v8bf*)&xAg[16 + M][c * 32 + kb8],
                        *(const v8bf*)&xAg[16 + M][c * 32 + 16 + kb8]);
    }

    for (int jt = 0; jt < 32; ++jt) {
        const int no = wave * 512 + jt * 16;
        const __bf16* brow = PRE ? (Bws + (size_t)(no + nn) * RT) : nullptr;

        v8f acc2a = {}, acc2b = {};
        #pragma unroll
        for (int c = 0; c < 4; ++c) {
            v16bf b2;
            const int Kst = c * 32 + ko16;
            if constexpr (PRE) {
                v8bf lo = *(const v8bf*)(brow + Kst);
                v8bf hi = *(const v8bf*)(brow + Kst + 16);
                b2 = cat8(lo, hi);
            } else {
                #pragma unroll
                for (int e = 0; e < 16; ++e)
                    b2[e] = f2bf(lora_B[(size_t)(Kst + e + (e & 8)) * OUT_DIM + no + nn]);
            }
            // one B operand feeds both row tiles
            acc2a = __builtin_amdgcn_wmma_f32_16x16x32_bf16(
                false, a2[0][c], false, b2, (short)0, acc2a, false, false);
            acc2b = __builtin_amdgcn_wmma_f32_16x16x32_bf16(
                false, a2[1][c], false, b2, (short)0, acc2b, false, false);
        }
        #pragma unroll
        for (int r = 0; r < 8; ++r) {
            out[(size_t)(row0 + Mb + r) * OUT_DIM + no + nn]      = acc2a[r];
            out[(size_t)(row0 + 16 + Mb + r) * OUT_DIM + no + nn] = acc2b[r];
        }
    }
}

extern "C" void kernel_launch(void* const* d_in, const int* in_sizes, int n_in,
                              void* d_out, int out_size, void* d_ws, size_t ws_size,
                              hipStream_t stream) {
    (void)in_sizes; (void)n_in; (void)out_size;
    const float* x      = (const float*)d_in[0];
    const float* lora_A = (const float*)d_in[1];
    const float* lora_B = (const float*)d_in[2];
    const int*   tid    = (const int*)d_in[3];
    float*       out    = (float*)d_out;

    dim3 grid(BATCH / 32);
    dim3 block(256);

    if (d_ws && ws_size >= WS_NEEDED) {
        __bf16* Aws = (__bf16*)d_ws;
        __bf16* Bws = Aws + AWS_ELEMS;
        preconv_A<<<dim3((unsigned)(AWS_ELEMS / 256)), block, 0, stream>>>(lora_A, Aws);
        preconv_B<<<dim3((unsigned)(BWS_ELEMS / 256)), block, 0, stream>>>(lora_B, Bws);
        id_gated_lora_kernel<true><<<grid, block, 0, stream>>>(
            x, lora_A, lora_B, tid, Aws, Bws, out);
    } else {
        id_gated_lora_kernel<false><<<grid, block, 0, stream>>>(
            x, lora_A, lora_B, tid, nullptr, nullptr, out);
    }
}
